// Block_22832046145821
// MI455X (gfx1250) — compile-verified, matrix operand order checked by reading, not run on hardware
//
#include <hip/hip_runtime.h>
#include <hip/hip_bf16.h>
#include <stdint.h>

// ---------------------------------------------------------------------------
// Problem constants (from reference): B=256, D0=16, D1=32, D2=64, D_FFN=2048
// ---------------------------------------------------------------------------
#define BATCH 256
#define DIM0  16
#define DIM1  32
#define DIM2  64
#define DFFN  2048
#define FF1   2048   // D1*D2
#define FF2   1024   // D0*D2

typedef __attribute__((ext_vector_type(16))) __bf16 v16bf;
typedef __attribute__((ext_vector_type(8)))  float  v8f;
typedef __attribute__((ext_vector_type(4)))  int    v4i_;

#define AS_GLOBAL __attribute__((address_space(1)))
#define AS_LOCAL  __attribute__((address_space(3)))

#if __has_builtin(__builtin_amdgcn_global_load_async_to_lds_b128) && \
    __has_builtin(__builtin_amdgcn_s_wait_asynccnt)
#define USE_ASYNC_COPY 1
#else
#define USE_ASYNC_COPY 0
#endif

// Cheap float->bf16 (round-half-up) and packed pair via v_perm_b32.
__device__ __forceinline__ unsigned short f2bf(float f) {
  return (unsigned short)((__float_as_uint(f) + 0x8000u) >> 16);
}
__device__ __forceinline__ unsigned int pack2bf(float lo, float hi) {
  unsigned int a = __float_as_uint(lo) + 0x8000u;
  unsigned int b = __float_as_uint(hi) + 0x8000u;
  // result bytes: [lo.b2, lo.b3, hi.b2, hi.b3]  (src1 supplies bytes 0-3)
  return __builtin_amdgcn_perm(b, a, 0x07060302u);
}

__device__ __forceinline__ float gelu_exact(float v) {
  return 0.5f * v * (1.0f + erff(v * 0.70710678118654752f));
}

// ---------------------------------------------------------------------------
// Convert x [b][i][j][d2] f32 -> two group-major bf16 activation layouts:
//   X1bf [i][b][j*64+d2]  (16 groups, K=2048 contiguous)
//   X2bf [j][b][i*64+d2]  (32 groups, K=1024 contiguous)
// ---------------------------------------------------------------------------
__global__ void convert_x_kernel(const float* __restrict__ x,
                                 unsigned short* __restrict__ x1bf,
                                 unsigned short* __restrict__ x2bf) {
  int idx = blockIdx.x * blockDim.x + threadIdx.x;  // < 256*16*32*64
  int d2 = idx & 63;
  int j  = (idx >> 6) & 31;
  int i  = (idx >> 11) & 15;
  int b  = idx >> 15;
  unsigned short v = f2bf(x[idx]);
  x1bf[((size_t)i * BATCH + b) * FF1 + j * DIM2 + d2] = v;
  x2bf[((size_t)j * BATCH + b) * FF2 + i * DIM2 + d2] = v;
}

// ---------------------------------------------------------------------------
// Grouped GEMM:  out[g][m][n] = act( sum_k A[g][m][k] * W[g][n][k] + bias[g][n] )
//   A: bf16 [G][256][K] row-major (K contiguous)  -> LDS via async copy
//   W: f32  [G][N][K]   row-major (K contiguous)  -> bf16 in LDS (v_perm pack)
// Block tile: 256(M) x 64(N) x 32(K): whole batch per block so every weight
// element is streamed from HBM exactly once (weight-BW-bound roofline).
// 256 threads = 8 waves in a 4x2 wave grid; each wave owns a 64x32 tile =
// 4x2 v_wmma_f32_16x16x32_bf16 tiles (64 accum VGPRs). LDS double-buffered.
// ---------------------------------------------------------------------------
template <bool GELU_BF16_OUT>
__global__ __launch_bounds__(256)
void grouped_gemm_wmma(const unsigned short* __restrict__ A,
                       const float* __restrict__ W,
                       const float* __restrict__ bias,
                       void* __restrict__ out,
                       int N, int K,
                       size_t outGStride, size_t outRowStride) {
  __shared__ unsigned short As[2][256 * 32];  // 2 x 16 KB
  __shared__ unsigned short Ws[2][64 * 32];   // 2 x  4 KB

  const int g    = blockIdx.x;
  const int n0   = blockIdx.y * 64;
  const int tid  = threadIdx.x;
  const int lane = tid & 31;
  const int wave = tid >> 5;
  const int waveM = wave & 3;   // 4 row groups of 64
  const int waveN = wave >> 2;  // 2 col groups of 32

  const unsigned short* Ag = A + (size_t)g * BATCH * K;
  const float*          Wg = W + ((size_t)g * N + n0) * K;

  // staging coordinates: A tile = 1024 16B chunks (4/thread), W = 8 f32/thread
  const int wRow = (tid * 8) >> 5;
  const int wKK  = (tid * 8) & 31;

  v8f acc[4][2];
  {
    v8f z = {0.f, 0.f, 0.f, 0.f, 0.f, 0.f, 0.f, 0.f};
    #pragma unroll
    for (int a = 0; a < 4; ++a) { acc[a][0] = z; acc[a][1] = z; }
  }

  union Frag { uint4 u[2]; v16bf v; };
  const int kb = (lane >> 4) << 3;  // per-lane K sub-chunk base: 0 or 8

  const int ksteps = K >> 5;

#if !USE_ASYNC_COPY
  uint4 areg[4];
#endif
  float4 wf0, wf1;

  // ---------------- prologue: stage tile 0 into buffer 0 ----------------
  #pragma unroll
  for (int p = 0; p < 4; ++p) {
    int chunk = tid + p * 256;
    int row = chunk >> 2;
    int kk  = (chunk & 3) * 8;
#if USE_ASYNC_COPY
    __builtin_amdgcn_global_load_async_to_lds_b128(
        (AS_GLOBAL v4i_*)(Ag + (size_t)row * K + kk),
        (AS_LOCAL v4i_*)&As[0][row * 32 + kk], 0, 0);
#else
    areg[p] = *reinterpret_cast<const uint4*>(Ag + (size_t)row * K + kk);
#endif
  }
  {
    const float* wsrc = Wg + (size_t)wRow * K + wKK;
    wf0 = *reinterpret_cast<const float4*>(wsrc);
    wf1 = *reinterpret_cast<const float4*>(wsrc + 4);
    uint4 pack;
    pack.x = pack2bf(wf0.x, wf0.y);
    pack.y = pack2bf(wf0.z, wf0.w);
    pack.z = pack2bf(wf1.x, wf1.y);
    pack.w = pack2bf(wf1.z, wf1.w);
    *reinterpret_cast<uint4*>(&Ws[0][wRow * 32 + wKK]) = pack;
#if !USE_ASYNC_COPY
    #pragma unroll
    for (int p = 0; p < 4; ++p) {
      int chunk = tid + p * 256;
      *reinterpret_cast<uint4*>(&As[0][(chunk >> 2) * 32 + (chunk & 3) * 8]) = areg[p];
    }
#endif
  }
#if USE_ASYNC_COPY
  __builtin_amdgcn_s_wait_asynccnt(0);
#endif
  __syncthreads();

  // ---------------- main K loop, double buffered ----------------
  for (int ks = 0; ks < ksteps; ++ks) {
    const int cur = ks & 1;
    const int nxt = cur ^ 1;
    const int k0n = (ks + 1) << 5;
    const bool more = (ks + 1 < ksteps);

    if (more) {
      #pragma unroll
      for (int p = 0; p < 4; ++p) {
        int chunk = tid + p * 256;
        int row = chunk >> 2;
        int kk  = (chunk & 3) * 8;
#if USE_ASYNC_COPY
        __builtin_amdgcn_global_load_async_to_lds_b128(
            (AS_GLOBAL v4i_*)(Ag + (size_t)row * K + k0n + kk),
            (AS_LOCAL v4i_*)&As[nxt][row * 32 + kk], 0, 0);
#else
        areg[p] = *reinterpret_cast<const uint4*>(Ag + (size_t)row * K + k0n + kk);
#endif
      }
      const float* wsrc = Wg + (size_t)wRow * K + k0n + wKK;
      wf0 = *reinterpret_cast<const float4*>(wsrc);
      wf1 = *reinterpret_cast<const float4*>(wsrc + 4);
      if (ks + 2 < ksteps)
        __builtin_prefetch(Wg + (size_t)(tid & 63) * K + ((ks + 2) << 5), 0, 1);
    }

    // ---- compute current buffer: fragments per CDNA5 16-bit striping ----
    {
      const uint4* As4 = reinterpret_cast<const uint4*>(As[cur]);
      const uint4* Ws4 = reinterpret_cast<const uint4*>(Ws[cur]);
      Frag af[4], bfr[2];
      #pragma unroll
      for (int t = 0; t < 4; ++t) {
        int arow = waveM * 64 + t * 16 + (lane & 15);
        af[t].u[0] = As4[(arow * 32 + kb) >> 3];       // K = kb .. kb+7
        af[t].u[1] = As4[(arow * 32 + kb + 16) >> 3];  // K = kb+16 .. kb+23
      }
      #pragma unroll
      for (int t = 0; t < 2; ++t) {
        int bcol = waveN * 32 + t * 16 + (lane & 15);
        bfr[t].u[0] = Ws4[(bcol * 32 + kb) >> 3];
        bfr[t].u[1] = Ws4[(bcol * 32 + kb + 16) >> 3];
      }
      #pragma unroll
      for (int tm = 0; tm < 4; ++tm)
        #pragma unroll
        for (int tn = 0; tn < 2; ++tn)
          acc[tm][tn] = __builtin_amdgcn_wmma_f32_16x16x32_bf16(
              false, af[tm].v, false, bfr[tn].v,
              (short)0, acc[tm][tn], false, false);
    }

    if (more) {
      uint4 pack;
      pack.x = pack2bf(wf0.x, wf0.y);
      pack.y = pack2bf(wf0.z, wf0.w);
      pack.z = pack2bf(wf1.x, wf1.y);
      pack.w = pack2bf(wf1.z, wf1.w);
      *reinterpret_cast<uint4*>(&Ws[nxt][wRow * 32 + wKK]) = pack;
#if !USE_ASYNC_COPY
      #pragma unroll
      for (int p = 0; p < 4; ++p) {
        int chunk = tid + p * 256;
        *reinterpret_cast<uint4*>(&As[nxt][(chunk >> 2) * 32 + (chunk & 3) * 8]) = areg[p];
      }
#else
      __builtin_amdgcn_s_wait_asynccnt(0);
#endif
    }
    __syncthreads();
  }

  // ---- epilogue: C/D layout  row = v + (lane>>4)*8, col = lane&15 ----
  #pragma unroll
  for (int tm = 0; tm < 4; ++tm) {
    #pragma unroll
    for (int tn = 0; tn < 2; ++tn) {
      int colG = n0 + waveN * 32 + tn * 16 + (lane & 15);
      float bv = bias[(size_t)g * N + colG];
      #pragma unroll
      for (int v = 0; v < 8; ++v) {
        int rowG = waveM * 64 + tm * 16 + ((lane >> 4) << 3) + v;
        float val = acc[tm][tn][v] + bv;
        size_t oidx = (size_t)g * outGStride + (size_t)rowG * outRowStride + colG;
        if (GELU_BF16_OUT) {
          ((unsigned short*)out)[oidx] = f2bf(gelu_exact(val));
        } else {
          ((float*)out)[oidx] = val;
        }
      }
    }
  }
}

// ---------------------------------------------------------------------------
// out[b][i][j][d2] = x + 0.5*(u + v); u already in d_out ([b][i][f1] layout),
// v in workspace as [j][b][i*64+d2].
// ---------------------------------------------------------------------------
__global__ void combine_kernel(const float* __restrict__ x,
                               const float* __restrict__ Vbuf,
                               float* __restrict__ out) {
  int idx = blockIdx.x * blockDim.x + threadIdx.x;
  int d2 = idx & 63;
  int j  = (idx >> 6) & 31;
  int i  = (idx >> 11) & 15;
  int b  = idx >> 15;
  float u = out[idx];
  float v = Vbuf[((size_t)j * BATCH + b) * FF2 + i * DIM2 + d2];
  out[idx] = x[idx] + 0.5f * (u + v);
}

// ---------------------------------------------------------------------------
extern "C" void kernel_launch(void* const* d_in, const int* in_sizes, int n_in,
                              void* d_out, int out_size, void* d_ws, size_t ws_size,
                              hipStream_t stream) {
  const float* x    = (const float*)d_in[0];
  const float* W0_1 = (const float*)d_in[1];
  const float* b0_1 = (const float*)d_in[2];
  const float* W1_1 = (const float*)d_in[3];
  const float* b1_1 = (const float*)d_in[4];
  const float* W0_2 = (const float*)d_in[5];
  const float* b0_2 = (const float*)d_in[6];
  const float* W1_2 = (const float*)d_in[7];
  const float* b1_2 = (const float*)d_in[8];
  float* out = (float*)d_out;

  // Workspace layout (64 MB total, regions reused after last sequential read):
  //   [ 0,16) MB  X1bf  (read by GEMM1)   -- later overlapped by Vbuf
  //   [16,32) MB  X2bf  (read by GEMM3)   -- later overlapped by Vbuf
  //   [32,48) MB  H1bf  (read by GEMM2)   -- later overlapped by H2bf
  //   [32,64) MB  H2bf  (written GEMM3, read GEMM4)
  //   [ 0,32) MB  Vbuf  (written GEMM4, read combine)
  char* ws = (char*)d_ws;
  unsigned short* X1bf = (unsigned short*)(ws);
  unsigned short* X2bf = (unsigned short*)(ws + ((size_t)16 << 20));
  unsigned short* H1bf = (unsigned short*)(ws + ((size_t)32 << 20));
  unsigned short* H2bf = (unsigned short*)(ws + ((size_t)32 << 20));
  float*          Vbuf = (float*)(ws);

  const int total = BATCH * DIM0 * DIM1 * DIM2;  // 8,388,608
  convert_x_kernel<<<total / 256, 256, 0, stream>>>(x, X1bf, X2bf);

  // mlp1: h1 = gelu(x1 @ W0_1[i]^T + b0_1)        [16][256][2048] bf16
  grouped_gemm_wmma<true><<<dim3(DIM0, DFFN / 64), 256, 0, stream>>>(
      X1bf, W0_1, b0_1, H1bf, DFFN, FF1,
      (size_t)BATCH * DFFN, (size_t)DFFN);

  // u = h1 @ W1_1[i]^T + b1_1  -> straight into d_out as [b][i][f1]
  grouped_gemm_wmma<false><<<dim3(DIM0, FF1 / 64), 256, 0, stream>>>(
      H1bf, W1_1, b1_1, out, FF1, DFFN,
      (size_t)FF1, (size_t)DIM0 * FF1);

  // mlp2: h2 = gelu(x2 @ W0_2[j]^T + b0_2)        [32][256][2048] bf16
  grouped_gemm_wmma<true><<<dim3(DIM1, DFFN / 64), 256, 0, stream>>>(
      X2bf, W0_2, b0_2, H2bf, DFFN, FF2,
      (size_t)BATCH * DFFN, (size_t)DFFN);

  // v = h2 @ W1_2[j]^T + b1_2  -> workspace [j][b][f2]
  grouped_gemm_wmma<false><<<dim3(DIM1, FF2 / 64), 256, 0, stream>>>(
      H2bf, W1_2, b1_2, Vbuf, FF2, DFFN,
      (size_t)BATCH * FF2, (size_t)FF2);

  // out = x + 0.5*(u + v)
  combine_kernel<<<total / 256, 256, 0, stream>>>(x, Vbuf, out);
}
